// LSTMTagger_54915451846910
// MI455X (gfx1250) — compile-verified, matrix operand order checked by reading, not run on hardware
//
#include <hip/hip_runtime.h>
#include <hip/hip_bf16.h>
#include <math.h>

// ---------------- problem constants (from reference) ----------------
#define S_  4096
#define L_  16
#define H_  512
#define HH_ 256     // H/2
#define EC_ 128
#define EW_ 512
#define KC_ 640     // EC + H  (char concat GEMM K)
#define NC_ 2048    // 4*H     (char gates)
#define KW_ 1024    // EW + H  (word input GEMM K)
#define NW_ 1024    // 4*HH
#define NPOS_ 64
#define NNER_ 32
#define NPAD_ 128   // projection weights padded to one block tile

typedef _Float16 v16h __attribute__((ext_vector_type(16)));
typedef _Float16 v8h  __attribute__((ext_vector_type(8)));
typedef float    v8f  __attribute__((ext_vector_type(8)));

__device__ __forceinline__ float sigf_(float x) { return 1.0f / (1.0f + __expf(-x)); }

// ---------------- WMMA 16x32 f16 fragment load (ISA layout) ----------
// lane 0..15 : row = lane,    K 0..7 (vgpr0-3) and 16..23 (vgpr4-7)
// lane 16..31: row = lane-16, K 8..15           and 24..31
// p already includes the per-lane row*ldk + (lane&16 ? 8 : 0) offset.
__device__ __forceinline__ v16h frag_at(const _Float16* p) {
  union { v16h v; v8h h[2]; } u;
  u.h[0] = *(const v8h*)(p);
  u.h[1] = *(const v8h*)(p + 16);
  return u.v;
}

// ---------------- fp16 WMMA GEMM:  C[M,N] = A[M,K] * B[Nb,K]^T + b1 + b2 -------
// A row-major lda=K, B row-major (weight layout, Nb >= N rows allocated) ldb=K,
// C row-major ldc=N (fp32). Block: 256 threads = 8 waves; block tile 128(M) x
// 128(N); wave tile 32(M) x 64(N) = 8 WMMAs per 32-wide K step, no load guards
// (B must be padded up to the block tile in N).
__global__ __launch_bounds__(256)
void k_gemm_f16(const _Float16* __restrict__ A, const _Float16* __restrict__ B,
                const float* __restrict__ bias1, const float* __restrict__ bias2,
                float* __restrict__ C, int M, int N, int K) {
  int wave = threadIdx.x >> 5;
  int lane = threadIdx.x & 31;
  int wm = wave & 3;        // 0..3  -> M sub-tile (32 rows each)
  int wn = wave >> 2;       // 0..1  -> N sub-tile (64 cols each)
  int m0 = blockIdx.y * 128 + wm * 32;
  int n0 = blockIdx.x * 128 + wn * 64;

  int rsel = lane & 15;
  int ksel = (lane & 16) ? 8 : 0;

  const _Float16* pa0 = A + (size_t)(m0 + rsel) * K + ksel;
  const _Float16* pa1 = pa0 + (size_t)16 * K;
  const _Float16* pb0 = B + (size_t)(n0 + rsel) * K + ksel;
  const _Float16* pb1 = pb0 + (size_t)16 * K;
  const _Float16* pb2 = pb0 + (size_t)32 * K;
  const _Float16* pb3 = pb0 + (size_t)48 * K;

  v8f acc[2][4] = {};

  for (int k = 0; k < K; k += 32) {
    v16h a0 = frag_at(pa0 + k);
    v16h a1 = frag_at(pa1 + k);
    v16h b0 = frag_at(pb0 + k);
    v16h b1 = frag_at(pb1 + k);
    v16h b2 = frag_at(pb2 + k);
    v16h b3 = frag_at(pb3 + k);
    acc[0][0] = __builtin_amdgcn_wmma_f32_16x16x32_f16(false, a0, false, b0, (short)0, acc[0][0], false, false);
    acc[0][1] = __builtin_amdgcn_wmma_f32_16x16x32_f16(false, a0, false, b1, (short)0, acc[0][1], false, false);
    acc[0][2] = __builtin_amdgcn_wmma_f32_16x16x32_f16(false, a0, false, b2, (short)0, acc[0][2], false, false);
    acc[0][3] = __builtin_amdgcn_wmma_f32_16x16x32_f16(false, a0, false, b3, (short)0, acc[0][3], false, false);
    acc[1][0] = __builtin_amdgcn_wmma_f32_16x16x32_f16(false, a1, false, b0, (short)0, acc[1][0], false, false);
    acc[1][1] = __builtin_amdgcn_wmma_f32_16x16x32_f16(false, a1, false, b1, (short)0, acc[1][1], false, false);
    acc[1][2] = __builtin_amdgcn_wmma_f32_16x16x32_f16(false, a1, false, b2, (short)0, acc[1][2], false, false);
    acc[1][3] = __builtin_amdgcn_wmma_f32_16x16x32_f16(false, a1, false, b3, (short)0, acc[1][3], false, false);
  }

  // C/D layout: lane = n + (m>=8 ? 16 : 0), vgpr r = m & 7
  int nlo = lane & 15;
  int mhi = (lane >> 4) * 8;
#pragma unroll
  for (int mt = 0; mt < 2; ++mt) {
#pragma unroll
    for (int nt = 0; nt < 4; ++nt) {
      int n = n0 + nt * 16 + nlo;
      if (n < N) {
        float b = (bias1 ? bias1[n] : 0.0f) + (bias2 ? bias2[n] : 0.0f);
#pragma unroll
        for (int r = 0; r < 8; ++r) {
          int m = m0 + mt * 16 + mhi + r;
          C[(size_t)m * N + n] = acc[mt][nt][r] + b;
        }
      }
    }
  }
}

// ---------------- small utility kernels ----------------
__global__ __launch_bounds__(256)
void k_cvt_f16(const float* __restrict__ src, _Float16* __restrict__ dst, int n) {
  int i = blockIdx.x * 256 + threadIdx.x;
  if (i < n) dst[i] = (_Float16)src[i];
}

// convert [rows, cols] fp32 -> fp16 padded to [padRows, cols] with zeros
__global__ __launch_bounds__(256)
void k_cvt_f16_pad(const float* __restrict__ src, _Float16* __restrict__ dst,
                   int rows, int cols, int padRows) {
  int i = blockIdx.x * 256 + threadIdx.x;
  if (i >= padRows * cols) return;
  int r = i / cols;
  dst[i] = (r < rows) ? (_Float16)src[i] : (_Float16)0.0f;
}

// W_cat_c[n, 0:128] = Wih_c[n, :], W_cat_c[n, 128:640] = Whh_c[n, :]
__global__ __launch_bounds__(256)
void k_build_wcat(const float* __restrict__ Wih, const float* __restrict__ Whh,
                  _Float16* __restrict__ Wcat) {
  int i = blockIdx.x * 256 + threadIdx.x;
  if (i >= NC_ * KC_) return;
  int n = i / KC_, k = i - n * KC_;
  float v = (k < EC_) ? Wih[n * EC_ + k] : Whh[n * H_ + (k - EC_)];
  Wcat[i] = (_Float16)v;
}

// h0 = prefix_emb[feat_seq], c0 = 0; also seed fp16 h into char GEMM activations
__global__ __launch_bounds__(256)
void k_init_h0(const int* __restrict__ feat_seq, const float* __restrict__ prefix_emb,
               float* __restrict__ h, float* __restrict__ c, _Float16* __restrict__ Ac) {
  int i = blockIdx.x * 256 + threadIdx.x;
  if (i >= S_ * H_) return;
  int s = i >> 9, j = i & (H_ - 1);
  float v = prefix_emb[(size_t)feat_seq[s] * H_ + j];
  h[i] = v; c[i] = 0.0f;
  Ac[(size_t)s * KC_ + EC_ + j] = (_Float16)v;
}

// A_c[s, 0:128] = f16(char_emb[chars[s, t]]) for step t
__global__ __launch_bounds__(256)
void k_gather_chars(const int* __restrict__ chars, const float* __restrict__ char_emb,
                    _Float16* __restrict__ Ac, int t) {
  int i = blockIdx.x * 256 + threadIdx.x;
  if (i >= S_ * EC_) return;
  int s = i >> 7, k = i & (EC_ - 1);
  int ch = chars[s * L_ + t];
  Ac[(size_t)s * KC_ + k] = (_Float16)char_emb[(size_t)ch * EC_ + k];
}

// masked LSTM cell update for char LSTM; refresh fp16 h in activations
__global__ __launch_bounds__(256)
void k_char_cell(const float* __restrict__ gates, const int* __restrict__ char_lens,
                 float* __restrict__ h, float* __restrict__ c,
                 _Float16* __restrict__ Ac, int t) {
  int i = blockIdx.x * 256 + threadIdx.x;
  if (i >= S_ * H_) return;
  int s = i >> 9, j = i & (H_ - 1);
  const float* g = gates + (size_t)s * NC_;
  float gi = g[j], gf = g[j + H_], gg = g[j + 2 * H_], go = g[j + 3 * H_];
  float cn = sigf_(gf) * c[i] + sigf_(gi) * tanhf(gg);
  float hn = sigf_(go) * tanhf(cn);
  if (t < char_lens[s]) {
    c[i] = cn; h[i] = hn;
    Ac[(size_t)s * KC_ + EC_ + j] = (_Float16)hn;
  }
}

// wc[s, 0:512] = f16(word_emb[word_seq[s]]); wc[s, 512:1024] = f16(char_feat = h)
__global__ __launch_bounds__(256)
void k_build_wc(const int* __restrict__ word_seq, const float* __restrict__ word_emb,
                const float* __restrict__ h, _Float16* __restrict__ wc) {
  int i = blockIdx.x * 256 + threadIdx.x;
  if (i >= S_ * KW_) return;
  int s = i >> 10, k = i & (KW_ - 1);
  float v = (k < EW_) ? word_emb[(size_t)word_seq[s] * EW_ + k]
                      : h[(size_t)s * H_ + (k - EW_)];
  wc[i] = (_Float16)v;
}

// ---------------- persistent bidirectional word LSTM recurrence ----------------
// grid = 2 blocks (dir 0 = forward, 1 = backward), 1024 threads each.
// Input gate pre-sums G[s,1024] already hold x@Wih^T + bih + bhh.
// Each step: thread n adds h@Whh[n,:]^T (h in LDS, Whh fp16 served from L2),
// then threads 0..255 run the cell and publish new h.
__global__ __launch_bounds__(1024)
void k_word_lstm(const float* __restrict__ Gf, const float* __restrict__ Gb,
                 const _Float16* __restrict__ Whhf, const _Float16* __restrict__ Whhb,
                 _Float16* __restrict__ out16) {
  int dir = blockIdx.x;
  const float*     G = dir ? Gb   : Gf;
  const _Float16*  W = dir ? Whhb : Whhf;
  int colOff = dir ? HH_ : 0;

  __shared__ __align__(16) _Float16 h_sh[HH_];
  __shared__ float g_sh[NW_];

  int tid = threadIdx.x;
  if (tid < HH_) h_sh[tid] = (_Float16)0.0f;
  float creg = 0.0f;
  __syncthreads();

  const v8h* wr = (const v8h*)(W + (size_t)tid * HH_);
  const v8h* hr = (const v8h*)h_sh;

  for (int step = 0; step < S_; ++step) {
    int s = dir ? (S_ - 1 - step) : step;
    float acc = G[(size_t)s * NW_ + tid];
#pragma unroll 4
    for (int kk = 0; kk < HH_ / 8; ++kk) {
      v8h w = wr[kk];
      v8h hh = hr[kk];
#pragma unroll
      for (int e = 0; e < 8; ++e) acc += (float)w[e] * (float)hh[e];
    }
    g_sh[tid] = acc;
    __syncthreads();
    if (tid < HH_) {
      float gi = g_sh[tid], gf = g_sh[tid + HH_], gg = g_sh[tid + 2 * HH_], go = g_sh[tid + 3 * HH_];
      creg = sigf_(gf) * creg + sigf_(gi) * tanhf(gg);
      float hn = sigf_(go) * tanhf(creg);
      h_sh[tid] = (_Float16)hn;
      out16[(size_t)s * H_ + colOff + tid] = (_Float16)hn;
    }
    __syncthreads();
  }
}

// ---------------- in-place log_softmax over pos(64) and ner(32) rows -----------
__global__ __launch_bounds__(64)
void k_logsoftmax(float* __restrict__ pos, float* __restrict__ ner) {
  int r = blockIdx.x;
  int tid = threadIdx.x;
  __shared__ float red[64];

  // POS row (64 wide)
  float* rowp = pos + (size_t)r * NPOS_;
  float v = rowp[tid];
  red[tid] = v; __syncthreads();
  for (int st = 32; st > 0; st >>= 1) { if (tid < st) red[tid] = fmaxf(red[tid], red[tid + st]); __syncthreads(); }
  float mx = red[0]; __syncthreads();
  red[tid] = __expf(v - mx); __syncthreads();
  for (int st = 32; st > 0; st >>= 1) { if (tid < st) red[tid] += red[tid + st]; __syncthreads(); }
  float lse = logf(red[0]) + mx; __syncthreads();
  rowp[tid] = v - lse;
  __syncthreads();

  // NER row (32 wide), threads 0..31 hold data; all threads hit barriers
  float* rown = ner + (size_t)r * NNER_;
  float v2 = (tid < NNER_) ? rown[tid] : -1e30f;
  if (tid < NNER_) red[tid] = v2;
  __syncthreads();
  for (int st = 16; st > 0; st >>= 1) { if (tid < st) red[tid] = fmaxf(red[tid], red[tid + st]); __syncthreads(); }
  float mx2 = red[0]; __syncthreads();
  if (tid < NNER_) red[tid] = __expf(v2 - mx2);
  __syncthreads();
  for (int st = 16; st > 0; st >>= 1) { if (tid < st) red[tid] += red[tid + st]; __syncthreads(); }
  float lse2 = logf(red[0]) + mx2; __syncthreads();
  if (tid < NNER_) rown[tid] = v2 - lse2;
}

// ---------------- host-side orchestration ----------------
static inline dim3 gsz(int n) { return dim3((n + 255) / 256); }

extern "C" void kernel_launch(void* const* d_in, const int* in_sizes, int n_in,
                              void* d_out, int out_size, void* d_ws, size_t ws_size,
                              hipStream_t stream) {
  // inputs (setup_inputs order)
  const int*   word_seq   = (const int*)d_in[0];
  const int*   chars      = (const int*)d_in[1];
  const int*   char_lens  = (const int*)d_in[2];
  const int*   feat_seq   = (const int*)d_in[3];
  const float* char_emb   = (const float*)d_in[4];
  const float* word_emb   = (const float*)d_in[5];
  const float* prefix_emb = (const float*)d_in[6];
  const float* Wih_c = (const float*)d_in[7];
  const float* Whh_c = (const float*)d_in[8];
  const float* bih_c = (const float*)d_in[9];
  const float* bhh_c = (const float*)d_in[10];
  const float* Wih_f = (const float*)d_in[11];
  const float* Whh_f = (const float*)d_in[12];
  const float* bih_f = (const float*)d_in[13];
  const float* bhh_f = (const float*)d_in[14];
  const float* Wih_b = (const float*)d_in[15];
  const float* Whh_b = (const float*)d_in[16];
  const float* bih_b = (const float*)d_in[17];
  const float* bhh_b = (const float*)d_in[18];
  const float* Wpos  = (const float*)d_in[19];
  const float* bpos  = (const float*)d_in[20];
  const float* Wner  = (const float*)d_in[21];
  const float* bner  = (const float*)d_in[22];

  // workspace carve-up
  size_t off = 0;
  auto alloc = [&](size_t bytes) -> char* {
    char* p = (char*)d_ws + off;
    off += (bytes + 255) & ~(size_t)255;
    return p;
  };
  _Float16* Ac     = (_Float16*)alloc((size_t)S_ * KC_ * 2);   // char GEMM activations [S,640]
  _Float16* Wcat   = (_Float16*)alloc((size_t)NC_ * KC_ * 2);  // [2048,640]
  float*    gates  = (float*)   alloc((size_t)S_ * NC_ * 4);   // [S,2048]
  float*    hbuf   = (float*)   alloc((size_t)S_ * H_ * 4);
  float*    cbuf   = (float*)   alloc((size_t)S_ * H_ * 4);
  _Float16* wc16   = (_Float16*)alloc((size_t)S_ * KW_ * 2);   // [S,1024]
  _Float16* Wihf16 = (_Float16*)alloc((size_t)NW_ * KW_ * 2);
  _Float16* Wihb16 = (_Float16*)alloc((size_t)NW_ * KW_ * 2);
  _Float16* Whhf16 = (_Float16*)alloc((size_t)NW_ * HH_ * 2);
  _Float16* Whhb16 = (_Float16*)alloc((size_t)NW_ * HH_ * 2);
  float*    Gf     = (float*)   alloc((size_t)S_ * NW_ * 4);
  float*    Gb     = (float*)   alloc((size_t)S_ * NW_ * 4);
  _Float16* out16  = (_Float16*)alloc((size_t)S_ * H_ * 2);
  _Float16* Wpos16 = (_Float16*)alloc((size_t)NPAD_ * H_ * 2); // zero-padded to 128 rows
  _Float16* Wner16 = (_Float16*)alloc((size_t)NPAD_ * H_ * 2); // zero-padded to 128 rows
  (void)ws_size; (void)in_sizes; (void)n_in; (void)out_size;

  // ---- weight conversions (fp32 -> fp16) ----
  k_build_wcat<<<gsz(NC_ * KC_), 256, 0, stream>>>(Wih_c, Whh_c, Wcat);
  k_cvt_f16<<<gsz(NW_ * KW_), 256, 0, stream>>>(Wih_f, Wihf16, NW_ * KW_);
  k_cvt_f16<<<gsz(NW_ * KW_), 256, 0, stream>>>(Wih_b, Wihb16, NW_ * KW_);
  k_cvt_f16<<<gsz(NW_ * HH_), 256, 0, stream>>>(Whh_f, Whhf16, NW_ * HH_);
  k_cvt_f16<<<gsz(NW_ * HH_), 256, 0, stream>>>(Whh_b, Whhb16, NW_ * HH_);
  k_cvt_f16_pad<<<gsz(NPAD_ * H_), 256, 0, stream>>>(Wpos, Wpos16, NPOS_, H_, NPAD_);
  k_cvt_f16_pad<<<gsz(NPAD_ * H_), 256, 0, stream>>>(Wner, Wner16, NNER_, H_, NPAD_);

  // ---- char LSTM: h0 = prefix_emb[feat_seq], c0 = 0 ----
  k_init_h0<<<gsz(S_ * H_), 256, 0, stream>>>(feat_seq, prefix_emb, hbuf, cbuf, Ac);

  dim3 gemm_blk(256);
  dim3 grid_char(NC_ / 128, S_ / 128);   // (16, 32)
  for (int t = 0; t < L_; ++t) {
    k_gather_chars<<<gsz(S_ * EC_), 256, 0, stream>>>(chars, char_emb, Ac, t);
    k_gemm_f16<<<grid_char, gemm_blk, 0, stream>>>(Ac, Wcat, bih_c, bhh_c, gates, S_, NC_, KC_);
    k_char_cell<<<gsz(S_ * H_), 256, 0, stream>>>(gates, char_lens, hbuf, cbuf, Ac, t);
  }

  // ---- word inputs: wc = [word_emb | char_feat], precompute input gates ----
  k_build_wc<<<gsz(S_ * KW_), 256, 0, stream>>>(word_seq, word_emb, hbuf, wc16);
  dim3 grid_word(NW_ / 128, S_ / 128);   // (8, 32)
  k_gemm_f16<<<grid_word, gemm_blk, 0, stream>>>(wc16, Wihf16, bih_f, bhh_f, Gf, S_, NW_, KW_);
  k_gemm_f16<<<grid_word, gemm_blk, 0, stream>>>(wc16, Wihb16, bih_b, bhh_b, Gb, S_, NW_, KW_);

  // ---- serial bidirectional recurrence (both directions concurrent) ----
  k_word_lstm<<<2, 1024, 0, stream>>>(Gf, Gb, Whhf16, Whhb16, out16);

  // ---- projections straight into d_out, then in-place log_softmax ----
  float* pos_out = (float*)d_out;                       // [4096, 64]
  float* ner_out = (float*)d_out + (size_t)S_ * NPOS_;  // [4096, 32]
  dim3 grid_proj(1, S_ / 128);
  k_gemm_f16<<<grid_proj, gemm_blk, 0, stream>>>(out16, Wpos16, bpos, nullptr, pos_out, S_, NPOS_, H_);
  k_gemm_f16<<<grid_proj, gemm_blk, 0, stream>>>(out16, Wner16, bner, nullptr, ner_out, S_, NNER_, H_);
  k_logsoftmax<<<S_, 64, 0, stream>>>(pos_out, ner_out);
}
